// DGCNN_GNN_Layers_79680233276343
// MI455X (gfx1250) — matrix operand order repaired and checked by reading, not compiled, not run on hardware
//
#include <hip/hip_runtime.h>
#include <hip/hip_bf16.h>
#include <cmath>

#define PNODES 256
#define BGRAPHS 512
#define FDIM 64

typedef __attribute__((ext_vector_type(16))) _Float16 v16h;
typedef __attribute__((ext_vector_type(8)))  _Float16 v8h;
typedef __attribute__((ext_vector_type(8)))  float    v8f;

// ---------------------------------------------------------------------------
// Kernel 1: adj = I  (f32, float4 stores)
// ---------------------------------------------------------------------------
__global__ __launch_bounds__(256) void init_adj_kernel(float* __restrict__ adj, long nvec4) {
  long i = (long)blockIdx.x * blockDim.x + threadIdx.x;
  if (i >= nvec4) return;
  long e = i << 2;                                   // first float index
  int local = (int)(e & (long)(PNODES * PNODES - 1));
  int row = local >> 8;
  int col = local & (PNODES - 1);
  float v[4] = {0.f, 0.f, 0.f, 0.f};
  int j = row - col;
  if (j >= 0 && j < 4) v[j] = 1.0f;                  // diagonal element in this quad
  *(float4*)(adj + e) = make_float4(v[0], v[1], v[2], v[3]);
}

// ---------------------------------------------------------------------------
// Kernel 2: scatter edge weights (same-graph edges only) with f32 atomics
// ---------------------------------------------------------------------------
__global__ __launch_bounds__(256) void scatter_kernel(const int* __restrict__ src,
                                                      const int* __restrict__ dst,
                                                      const int* __restrict__ batch,
                                                      const float* __restrict__ w,
                                                      float* __restrict__ adj, int E) {
  int e = blockIdx.x * blockDim.x + threadIdx.x;
  if (e >= E) return;
  int s = src[e], d = dst[e];
  int g = batch[s];
  if (batch[d] != g) return;
  int ls = s - g * PNODES;
  int ld = d - g * PNODES;
  ls = ls < 0 ? 0 : (ls > PNODES - 1 ? PNODES - 1 : ls);
  ld = ld < 0 ? 0 : (ld > PNODES - 1 ? PNODES - 1 : ld);
  atomicAdd(adj + ((long)g * PNODES + ls) * PNODES + ld, w[e]);
}

// ---------------------------------------------------------------------------
// Kernel 3: row degree -> reciprocal, fused f32->f16 adjacency conversion
// (touches every adj element exactly once anyway)
// ---------------------------------------------------------------------------
__global__ __launch_bounds__(256) void rinv_cvt_kernel(const float* __restrict__ adj,
                                                       float* __restrict__ rinv,
                                                       _Float16* __restrict__ adjh,
                                                       int nrows) {
  int r = blockIdx.x * blockDim.x + threadIdx.x;
  if (r >= nrows) return;
  const float4* rp = (const float4*)(adj + (long)r * PNODES);
  _Float16* hp = adjh + (long)r * PNODES;
  float s = 0.f;
#pragma unroll 4
  for (int i = 0; i < PNODES / 8; i++) {
    float4 a = rp[2 * i];
    float4 c = rp[2 * i + 1];
    s += a.x + a.y + a.z + a.w + c.x + c.y + c.z + c.w;
    v8h h;
    h[0] = (_Float16)a.x; h[1] = (_Float16)a.y; h[2] = (_Float16)a.z; h[3] = (_Float16)a.w;
    h[4] = (_Float16)c.x; h[5] = (_Float16)c.y; h[6] = (_Float16)c.z; h[7] = (_Float16)c.w;
    *(v8h*)(hp + 8 * i) = h;
  }
  rinv[r] = (s > 0.f) ? (1.0f / s) : 0.f;
}

// ---------------------------------------------------------------------------
// Kernel 4: x f32 -> f16
// ---------------------------------------------------------------------------
__global__ __launch_bounds__(256) void cvt_kernel(const float* __restrict__ x,
                                                  _Float16* __restrict__ h, long n) {
  long i = (long)blockIdx.x * blockDim.x + threadIdx.x;
  if (i < n) h[i] = (_Float16)x[i];
}

// ---------------------------------------------------------------------------
// WMMA operand helpers (CDNA5 16-bit A/B VGPR layouts, wave32)
// A 16x32: lanes 0-15 row M=lane, elems[0:8]=K(k0..k0+7), elems[8:16]=K(k0+16..+23)
//          lanes 16-31 same rows, K offset +8.
// B 32x16: lanes 0-15 col N=lane, K order same as A (lanes 16-31 K offset +8).
// ---------------------------------------------------------------------------
__device__ __forceinline__ v16h cat8(v8h lo, v8h hi) {
  v16h r;
#pragma unroll
  for (int i = 0; i < 8; i++) { r[i] = lo[i]; r[i + 8] = hi[i]; }
  return r;
}

// ---------------------------------------------------------------------------
// Kernel 5: one DGCNN layer. grid = (B, 2): blockIdx.x = graph,
// blockIdx.y selects 8 of the 16 row-tiles for phase 2.
//   phase 1: T = H @ W   (WMMA, stored transposed in LDS; duplicated per y-block)
//   phase 2: O = A @ T   (WMMA, A = f16 adjacency from global)
//   out = tanh(rinv * O + b)
// ---------------------------------------------------------------------------
template <int IN, int OUT, bool FINAL>
__global__ __launch_bounds__(256) void dgcnn_layer_kernel(
    const _Float16* __restrict__ h_in, const _Float16* __restrict__ adjh,
    const float* __restrict__ rinv, const float* __restrict__ W,
    const float* __restrict__ bias, _Float16* __restrict__ h_out,
    float* __restrict__ out_final) {
  constexpr int OUTP = (OUT + 15) & ~15;   // padded output width (16 or 32)
  constexpr int COLT = OUTP / 16;          // column tiles
  constexpr int HS = IN + 8;               // padded LDS strides (16B-aligned rows)
  constexpr int TS = PNODES + 8;

  __shared__ _Float16 Hs[PNODES][HS];      // input features, row-major
  __shared__ _Float16 Wt[OUTP][HS];        // W transposed (zero-padded cols)
  __shared__ _Float16 Tt[OUTP][TS];        // T = H@W, transposed
  __shared__ float rv[PNODES];
  __shared__ float bs[OUTP];

  const int b = blockIdx.x;
  const int tid = threadIdx.x;
  const int lane = tid & 31;
  const int wave = tid >> 5;
  const int lhalf = lane >> 4;             // 0 or 1
  const int l16 = lane & 15;

  // ---- stage H (f16 global -> LDS, 4B chunks) ----
  const size_t hbase = (size_t)b * PNODES * IN;
  const unsigned int* hsrc = (const unsigned int*)(h_in + hbase);
  for (int i = tid; i < PNODES * IN / 2; i += 256) {
    int row = (2 * i) / IN, col = (2 * i) % IN;
    *(unsigned int*)&Hs[row][col] = hsrc[i];
  }
  // ---- stage W transposed (f32 -> f16, zero pad) ----
  for (int i = tid; i < OUTP * IN; i += 256) {
    int n = i / IN, k = i % IN;
    Wt[n][k] = (n < OUT) ? (_Float16)W[(size_t)k * OUT + n] : (_Float16)0.0f;
  }
  if (tid < OUTP) bs[tid] = (tid < OUT) ? bias[tid] : 0.0f;
  if (tid < PNODES) rv[tid] = rinv[(size_t)b * PNODES + tid];
  __syncthreads();

  // ---- phase 1: T = H @ W, stored transposed (constant trip count) ----
#pragma unroll
  for (int tt = 0; tt < 2 * COLT; ++tt) {
    const int t = wave + tt * 8;           // 0 .. 16*COLT-1, wave-uniform
    const int ti = t / COLT, tj = t % COLT;
    const int m = ti * 16 + l16;
    const int n = tj * 16 + l16;
    v8f acc = {};
#pragma unroll
    for (int k0 = 0; k0 < IN; k0 += 32) {
      const int kA = k0 + (lhalf << 3);
      v16h a = cat8(*(const v8h*)&Hs[m][kA], *(const v8h*)&Hs[m][kA + 16]);
      const int kB = k0 + (lhalf << 3);
      v16h bm = cat8(*(const v8h*)&Wt[n][kB], *(const v8h*)&Wt[n][kB + 16]);
      acc = __builtin_amdgcn_wmma_f32_16x16x32_f16(false, a, false, bm,
                                                   (short)0, acc, false, false);
    }
    const int mb = ti * 16 + (lhalf << 3);
    v8h tv;
#pragma unroll
    for (int r = 0; r < 8; r++) tv[r] = (_Float16)acc[r];
    *(v8h*)&Tt[n][mb] = tv;                // transposed store, 16B aligned
  }
  __syncthreads();

  // ---- phase 2: O = A @ T over this block's 8 row-tiles ----
  const _Float16* adjb = adjh + (size_t)b * PNODES * PNODES;
  const int ti0 = blockIdx.y * 8;
#pragma unroll
  for (int tt = 0; tt < COLT; ++tt) {
    const int t = wave + tt * 8;           // 0 .. 8*COLT-1, wave-uniform
    const int ti = ti0 + t / COLT;
    const int tj = t % COLT;
    const int m = ti * 16 + l16;
    const int n = tj * 16 + l16;
    const _Float16* arow = adjb + (size_t)m * PNODES;
    v8f acc = {};
#pragma unroll
    for (int k0 = 0; k0 < PNODES; k0 += 32) {
      const int kA = k0 + (lhalf << 3);
      if (k0 + 32 < PNODES) __builtin_prefetch(arow + kA + 32, 0, 1);
      v16h a = cat8(*(const v8h*)(arow + kA), *(const v8h*)(arow + kA + 16));
      const int kB = k0 + (lhalf << 3);
      v16h bm = cat8(*(const v8h*)&Tt[n][kB], *(const v8h*)&Tt[n][kB + 16]);
      acc = __builtin_amdgcn_wmma_f32_16x16x32_f16(false, a, false, bm,
                                                   (short)0, acc, false, false);
    }
#pragma unroll
    for (int r = 0; r < 8; r++) {
      const int mr = ti * 16 + r + (lhalf << 3);
      float val = tanhf(rv[mr] * acc[r] + bs[n]);
      if (FINAL) {
        if (n == 0) out_final[(size_t)b * PNODES + mr] = val;
      } else {
        if (n < OUT) h_out[((size_t)b * PNODES + mr) * OUT + n] = (_Float16)val;
      }
    }
  }
}

// ---------------------------------------------------------------------------
// Launch
// ---------------------------------------------------------------------------
extern "C" void kernel_launch(void* const* d_in, const int* in_sizes, int n_in,
                              void* d_out, int out_size, void* d_ws, size_t ws_size,
                              hipStream_t stream) {
  const float* x     = (const float*)d_in[0];
  const int*   ei    = (const int*)d_in[1];
  const int*   batch = (const int*)d_in[2];
  const float* emask = (const float*)d_in[3];
  const float* W0 = (const float*)d_in[4];  const float* b0 = (const float*)d_in[5];
  const float* W1 = (const float*)d_in[6];  const float* b1 = (const float*)d_in[7];
  const float* W2 = (const float*)d_in[8];  const float* b2 = (const float*)d_in[9];
  const float* W3 = (const float*)d_in[10]; const float* b3 = (const float*)d_in[11];

  const int  E   = in_sizes[1] / 2;
  const long NP  = (long)BGRAPHS * PNODES;                 // 131072 nodes
  const long ADJ = (long)BGRAPHS * PNODES * PNODES;        // 33.5M elements

  float*     adj  = (float*)d_ws;                          // 134 MB f32 (build + degree)
  float*     rinv = adj + ADJ;                             // 512 KB
  _Float16*  adjh = (_Float16*)(rinv + NP);                // 67 MB f16 (GEMM operand)
  _Float16*  h0   = adjh + ADJ;                            // N x 64 f16
  _Float16*  h1   = h0 + NP * FDIM;                        // N x 32 f16
  _Float16*  h2   = h1 + NP * 32;                          // N x 32 f16

  const long nv4 = ADJ / 4;
  init_adj_kernel<<<(unsigned)((nv4 + 255) / 256), 256, 0, stream>>>(adj, nv4);
  scatter_kernel<<<(E + 255) / 256, 256, 0, stream>>>(ei, ei + E, batch, emask, adj, E);
  rinv_cvt_kernel<<<(unsigned)((NP + 255) / 256), 256, 0, stream>>>(adj, rinv, adjh, (int)NP);
  const long nx = NP * FDIM;
  cvt_kernel<<<(unsigned)((nx + 255) / 256), 256, 0, stream>>>(x, h0, nx);

  dim3 grid(BGRAPHS, 2);
  dgcnn_layer_kernel<64, 32, false><<<grid, 256, 0, stream>>>(h0, adjh, rinv, W0, b0, h1, nullptr);
  dgcnn_layer_kernel<32, 32, false><<<grid, 256, 0, stream>>>(h1, adjh, rinv, W1, b1, h2, nullptr);
  dgcnn_layer_kernel<32, 32, false><<<grid, 256, 0, stream>>>(h2, adjh, rinv, W2, b2, h1, nullptr);
  dgcnn_layer_kernel<32, 1, true><<<grid, 256, 0, stream>>>(h1, adjh, rinv, W3, b3, nullptr,
                                                            (float*)d_out);
}
